// LSTM_thetafunc_56719338111703
// MI455X (gfx1250) — matrix operand order repaired and checked by reading, not compiled, not run on hardware
//
#include <hip/hip_runtime.h>
#include <hip/hip_bf16.h>

typedef __attribute__((ext_vector_type(16))) __bf16 v16bf;
typedef __attribute__((ext_vector_type(8)))  float  v8f;

#define BB   256
#define TT   128
#define SS   1024
#define PLEN 64
#define G4   4096   // 4*S
#define K2   2048   // 2*S

union Frag { v16bf v; uint4 q[2]; };

// ---------------------------------------------------------------------------
// Fast device math: raw v_exp_f32 / v_rcp_f32 forms
// ---------------------------------------------------------------------------
__device__ __forceinline__ float sigmoid_f(float x) {
    return __builtin_amdgcn_rcpf(1.0f + __expf(-x));
}
__device__ __forceinline__ float tanh_f(float x) {
    float xc = fminf(fmaxf(x, -15.0f), 15.0f);
    float e  = __expf(2.0f * xc);
    return (e - 1.0f) * __builtin_amdgcn_rcpf(e + 1.0f);
}

// ---------------------------------------------------------------------------
// Pack x (B,T,S) fp32 -> xb (T,B,S) bf16
// ---------------------------------------------------------------------------
__global__ __launch_bounds__(256) void pack_x_kernel(const float* __restrict__ x,
                                                     __hip_bfloat16* __restrict__ xb) {
    long idx = (long)blockIdx.x * 256 + threadIdx.x;   // over T*B*S = 2^25
    int t = (int)(idx >> 18);            // B*S = 2^18
    int r = (int)(idx & ((1 << 18) - 1));
    int b = r >> 10;
    int s = r & 1023;
    xb[idx] = __float2bfloat16(x[((long)b * TT + t) * SS + s]);
}

// ---------------------------------------------------------------------------
// Pack Wcat = [W_ih ; W_hh] (4096 x 2048) into WMMA B-fragment layout (bf16):
//   offset = (((nb*64 + kt)*8 + ct)*32 + lane)*16 + e
//   nb: hidden block (32 units), kt: k/32, ct = gate*2 + hiddenSub,
//   lane holds col n = lane%16, k = kt*32 + (e/8)*16 + (lane/16)*8 + e%8
// ---------------------------------------------------------------------------
__global__ __launch_bounds__(256) void pack_w_kernel(const float* __restrict__ W_ih,
                                                     const float* __restrict__ W_hh,
                                                     __hip_bfloat16* __restrict__ Bp) {
    long idx = (long)blockIdx.x * 256 + threadIdx.x;   // over K2*G4 = 2^23
    int e  = (int)(idx        & 15);
    int L  = (int)((idx >> 4) & 31);
    int ct = (int)((idx >> 9) & 7);
    int kt = (int)((idx >> 12) & 63);
    int nb = (int)(idx >> 18);                          // 0..31
    int k    = kt * 32 + ((e >> 3) << 4) + ((L >> 4) << 3) + (e & 7); // 0..2047
    int gate = ct >> 1;
    int row  = gate * SS + nb * 32 + ((ct & 1) << 4) + (L & 15);
    float v  = (k < SS) ? W_ih[(long)row * SS + k]
                        : W_hh[(long)row * SS + (k - SS)];
    Bp[idx] = __float2bfloat16(v);
}

// ---------------------------------------------------------------------------
// bias = b_ih + b_hh ; h0 = 0 (bf16) ; c = 0 (fp32)
// ---------------------------------------------------------------------------
__global__ __launch_bounds__(256) void init_kernel(const float* __restrict__ b_ih,
                                                   const float* __restrict__ b_hh,
                                                   float* __restrict__ bias,
                                                   unsigned short* __restrict__ h0,
                                                   float* __restrict__ c) {
    int idx = blockIdx.x * 256 + threadIdx.x;           // over B*S = 262144
    c[idx]  = 0.0f;
    h0[idx] = 0;
    if (idx < G4) bias[idx] = b_ih[idx] + b_hh[idx];
}

// ---------------------------------------------------------------------------
// Pipeline stage load: A fragments for two m-tiles + B fragments for 4 gates.
// aoff is a *uniform* (scalar) element offset selecting x- vs h-part, so the
// per-lane address math is a single v_add of an SGPR pair.
// kt may reach 64 (branch-free prefetch); over-reads stay inside workspace.
// ---------------------------------------------------------------------------
__device__ __forceinline__ void load_stage(
    Frag& a0, Frag& a1, Frag& b0, Frag& b1, Frag& b2, Frag& b3,
    const __hip_bfloat16* __restrict__ ax,   // xpart + row_b*SS (per-lane base)
    long aoff,                                // scalar: kt*32 (+ hdelta if kt>=32)
    const __hip_bfloat16* __restrict__ bbase,// Bp + nb*64*4096 + hs*512 + L*16
    int kt, int base8)
{
    const __hip_bfloat16* p = ax + aoff;
    a0.q[0] = *(const uint4*)(p + base8);
    a0.q[1] = *(const uint4*)(p + 16 + base8);
    a1.q[0] = *(const uint4*)(p + 64 * SS + base8);      // second m-tile: +64 rows
    a1.q[1] = *(const uint4*)(p + 64 * SS + 16 + base8);

    const __hip_bfloat16* tb = bbase + (long)kt * 4096;  // hs folded into bbase
    const uint4* q;
    q = (const uint4*)(tb + 0 * 1024); b0.q[0] = q[0]; b0.q[1] = q[1];
    q = (const uint4*)(tb + 1 * 1024); b1.q[0] = q[0]; b1.q[1] = q[1];
    q = (const uint4*)(tb + 2 * 1024); b2.q[0] = q[0]; b2.q[1] = q[1];
    q = (const uint4*)(tb + 3 * 1024); b3.q[0] = q[0]; b3.q[1] = q[1];
    // Warm WGP$ (locality 3 -> WGP scope) for the tile 8 k-steps ahead.
    // Issue on even kt only; each prefetch covers the lane-strided 2KB region.
    if ((kt & 1) == 0)
        __builtin_prefetch((const void*)(tb + 8 * 4096), 0, 3);
}

#define WMMA_STAGE(A0, A1, B0, B1, B2, B3)                                                          \
    acc_i0 = __builtin_amdgcn_wmma_f32_16x16x32_bf16(false, A0.v, false, B0.v, (short)0, acc_i0, false, false); \
    acc_i1 = __builtin_amdgcn_wmma_f32_16x16x32_bf16(false, A1.v, false, B0.v, (short)0, acc_i1, false, false); \
    acc_f0 = __builtin_amdgcn_wmma_f32_16x16x32_bf16(false, A0.v, false, B1.v, (short)0, acc_f0, false, false); \
    acc_f1 = __builtin_amdgcn_wmma_f32_16x16x32_bf16(false, A1.v, false, B1.v, (short)0, acc_f1, false, false); \
    acc_g0 = __builtin_amdgcn_wmma_f32_16x16x32_bf16(false, A0.v, false, B2.v, (short)0, acc_g0, false, false); \
    acc_g1 = __builtin_amdgcn_wmma_f32_16x16x32_bf16(false, A1.v, false, B2.v, (short)0, acc_g1, false, false); \
    acc_o0 = __builtin_amdgcn_wmma_f32_16x16x32_bf16(false, A0.v, false, B3.v, (short)0, acc_o0, false, false); \
    acc_o1 = __builtin_amdgcn_wmma_f32_16x16x32_bf16(false, A1.v, false, B3.v, (short)0, acc_o1, false, false);

// ---------------------------------------------------------------------------
// One LSTM timestep, fully fused, software-pipelined (depth-1 reg double buf):
//   gates = [xpart | hin] @ Wcat^T  (bf16 WMMA, fp32 acc)
//   c' = sig(f)*c + sig(i)*tanh(g); h' = sig(o)*tanh(c')
// grid = (2, 32): blockIdx.x -> 128 batch rows, blockIdx.y -> 32 hidden units.
// 8 waves: wave w -> m-tiles {w&3, (w&3)+4}, hidden-sub (w>>2); each wave owns
// i,f,g,o for its two 16x16 (batch,hidden) tiles -> elementwise in-register.
// ---------------------------------------------------------------------------
__global__ __launch_bounds__(256) void lstm_step_kernel(
    const __hip_bfloat16* __restrict__ xpart,   // A cols [0,1024)
    const __hip_bfloat16* __restrict__ hin,     // A cols [1024,2048)
    const __hip_bfloat16* __restrict__ Bp,      // packed weights
    const float* __restrict__ bias,             // 4096
    float* __restrict__ cstate,                 // B*S fp32
    __hip_bfloat16* __restrict__ hout,          // B*S bf16
    float* __restrict__ dec_out,                // null for encoder
    int p)                                      // decoder step index
{
    const int tid    = threadIdx.x;
    const int L      = tid & 31;
    const int w      = tid >> 5;
    const int mt     = w & 3;          // first m-tile of this wave
    const int hs     = w >> 2;         // hidden sub-tile (0/1)
    const int m0     = blockIdx.x * 128;
    const int nb     = blockIdx.y;
    const int lane_m = L & 15;
    const int hi     = L >> 4;         // half-wave
    const int base8  = hi * 8;         // A-frag k sub-offset (in halves)

    const int row_b  = m0 + mt * 16 + lane_m;        // A row (m-tile set 0)
    const int n_lane = nb * 32 + hs * 16 + lane_m;   // hidden col of C

    const float bi = bias[n_lane];
    const float bf = bias[SS + n_lane];
    const float bg = bias[2 * SS + n_lane];
    const float bo = bias[3 * SS + n_lane];

    v8f acc_i0 = {}, acc_f0 = {}, acc_g0 = {}, acc_o0 = {};
    v8f acc_i1 = {}, acc_f1 = {}, acc_g1 = {}, acc_o1 = {};

    const __hip_bfloat16* ax = xpart + (long)row_b * SS;
    // Uniform scalar delta: for kt>=32, p = ax + kt*32 + hdelta == hin + row*SS + (kt-32)*32
    const long hdelta = ((long)((const char*)hin - (const char*)xpart) >> 1) - SS;
    // Fold hs into the B base: ct = gate*2 + hs, stride 512 halves -> +hs*512
    const __hip_bfloat16* bbase = Bp + (long)nb * 64 * 4096 + hs * 512 + L * 16;

    Frag aA0, aA1, bA0, bA1, bA2, bA3;   // stage P0
    Frag aB0, aB1, bB0, bB1, bB2, bB3;   // stage P1

    load_stage(aA0, aA1, bA0, bA1, bA2, bA3, ax, 0, bbase, 0, base8);

    for (int kt = 0; kt < 64; kt += 2) {
        {
            const int  k1 = kt + 1;
            const long o1 = (long)k1 * 32 + ((k1 >= 32) ? hdelta : 0);
            load_stage(aB0, aB1, bB0, bB1, bB2, bB3, ax, o1, bbase, k1, base8);
        }
        WMMA_STAGE(aA0, aA1, bA0, bA1, bA2, bA3)
        {
            const int  k2 = kt + 2;
            const long o2 = (long)k2 * 32 + ((k2 >= 32) ? hdelta : 0);
            load_stage(aA0, aA1, bA0, bA1, bA2, bA3, ax, o2, bbase, k2, base8);
        }
        WMMA_STAGE(aB0, aB1, bB0, bB1, bB2, bB3)
    }

    // Epilogue: C layout -> VGPR r holds (M = r + hi*8, N = lane_m)
    auto epilogue = [&](const v8f& Ai, const v8f& Af, const v8f& Ag, const v8f& Ao,
                        int mbase) {
        #pragma unroll
        for (int r = 0; r < 8; ++r) {
            const int  bb  = mbase + hi * 8 + r;
            const long off = (long)bb * SS + n_lane;
            float si = sigmoid_f(Ai[r] + bi);
            float sf = sigmoid_f(Af[r] + bf);
            float so = sigmoid_f(Ao[r] + bo);
            float tg = tanh_f(Ag[r] + bg);
            float cn = sf * cstate[off] + si * tg;
            cstate[off] = cn;
            float hn = so * tanh_f(cn);
            hout[off] = __float2bfloat16(hn);
            if (dec_out)
                dec_out[(long)bb * (PLEN * SS) + (long)p * SS + n_lane] = hn;
        }
    };
    epilogue(acc_i0, acc_f0, acc_g0, acc_o0, m0 + mt * 16);
    epilogue(acc_i1, acc_f1, acc_g1, acc_o1, m0 + mt * 16 + 64);
}

// ---------------------------------------------------------------------------
// Workspace layout (bytes):
//   xb   : 0          (T*B*S*2      = 67108864)
//   Bp   : 67108864   (K2*G4*2      = 16777216)
//   bias : 83886080   (4096*4       = 16384)
//   h0   : 83902464   (B*S*2        = 524288)
//   h1   : 84426752   (B*S*2        = 524288)
//   c    : 84951040   (B*S*4        = 1048576)   total ~86 MB
// ---------------------------------------------------------------------------
extern "C" void kernel_launch(void* const* d_in, const int* in_sizes, int n_in,
                              void* d_out, int out_size, void* d_ws, size_t ws_size,
                              hipStream_t stream) {
    const float* x    = (const float*)d_in[0];
    const float* W_ih = (const float*)d_in[1];
    const float* W_hh = (const float*)d_in[2];
    const float* b_ih = (const float*)d_in[3];
    const float* b_hh = (const float*)d_in[4];

    if (ws_size < (size_t)86000000) return;

    char* ws = (char*)d_ws;
    __hip_bfloat16* xb   = (__hip_bfloat16*)(ws + 0);
    __hip_bfloat16* Bp   = (__hip_bfloat16*)(ws + 67108864);
    float*          bias = (float*)(ws + 83886080);
    __hip_bfloat16* h0   = (__hip_bfloat16*)(ws + 83902464);
    __hip_bfloat16* h1   = (__hip_bfloat16*)(ws + 84426752);
    float*          cst  = (float*)(ws + 84951040);

    pack_x_kernel<<<(TT * BB * SS) / 256, 256, 0, stream>>>(x, xb);
    pack_w_kernel<<<(K2 * G4) / 256,     256, 0, stream>>>(W_ih, W_hh, Bp);
    init_kernel<<<(BB * SS) / 256,       256, 0, stream>>>(b_ih, b_hh, bias,
                                                           (unsigned short*)h0, cst);

    __hip_bfloat16* bufs[2] = { h0, h1 };
    dim3 grid(2, 32);
    int cur = 0;

    // Encoder: 128 steps over xb
    for (int t = 0; t < TT; ++t) {
        lstm_step_kernel<<<grid, 256, 0, stream>>>(
            xb + (long)t * (BB * SS), bufs[cur], Bp, bias, cst, bufs[cur ^ 1],
            nullptr, 0);
        cur ^= 1;
    }

    // Decoder: 64 steps, feed h back as the input; emit fp32 outputs
    float* out = (float*)d_out;
    for (int p = 0; p < PLEN; ++p) {
        lstm_step_kernel<<<grid, 256, 0, stream>>>(
            bufs[cur], bufs[cur], Bp, bias, cst, bufs[cur ^ 1],
            out, p);
        cur ^= 1;
    }
}
